// MultiFeatureNet_59339268161865
// MI455X (gfx1250) — matrix operand order repaired and checked by reading, not compiled, not run on hardware
//
#include <hip/hip_runtime.h>
#include <hip/hip_bf16.h>

typedef __attribute__((ext_vector_type(16))) _Float16 v16h;
typedef __attribute__((ext_vector_type(8)))  float    v8f;
typedef __attribute__((ext_vector_type(8)))  unsigned  v8u;

#define GN 100000
#define GE 1600000
#define GH 256
#define GG 64

#define BM 64           // rows per block
#define BK 32           // K per step
#define LROW 20         // dwords per LDS row (80B: 16B-aligned, bank-spread)

#if defined(__AMDGCN__) && \
    __has_builtin(__builtin_amdgcn_global_load_async_to_lds_b128) && \
    __has_builtin(__builtin_amdgcn_s_wait_asynccnt)
#define HAS_ASYNC_LDS 1
typedef __attribute__((__vector_size__(16)))  int      v4i;   // builtin's elem type
typedef __attribute__((address_space(3)))     v4i      as3_v4i;
#else
#define HAS_ASYNC_LDS 0
#endif

// ---------------------------------------------------------------------------
// f32 [rows,K] -> f16 [rows,Kp] with zero padding (coalesced, branchless)
// ---------------------------------------------------------------------------
__global__ void cast_pad_f16(const float* __restrict__ A, _Float16* __restrict__ O,
                             int K, int Kp, long long rows) {
    long long i = (long long)blockIdx.x * blockDim.x + threadIdx.x;
    long long total = rows * Kp;
    if (i >= total) return;
    long long r = i / Kp;
    int k = (int)(i - r * Kp);
    O[i] = (k < K) ? (_Float16)A[r * K + k] : (_Float16)0.0f;
}

// ---------------------------------------------------------------------------
// Weight convert + transpose + K-pad: W[K][256] f32 -> Wt[256][Kp] f16
// ---------------------------------------------------------------------------
__global__ void cvt_transpose(const float* __restrict__ W, _Float16* __restrict__ Wt,
                              int K, int Nc, int Kp) {
    long long i = (long long)blockIdx.x * blockDim.x + threadIdx.x;
    long long total = (long long)Nc * Kp;
    if (i >= total) return;
    int n = (int)(i / Kp);
    int k = (int)(i % Kp);
    Wt[i] = (k < K) ? (_Float16)W[(long long)k * Nc + n] : (_Float16)0.0f;
}

// ---------------------------------------------------------------------------
// WMMA GEMM: C[M x 256] = A[M x Kp](f16) * Wt^T   (Wt [256 x Kp] f16, Kp%32==0)
// block: 64 rows x 256 cols, 8 waves = 4(M) x 2(N), 8 wmma / wave / K-step.
// Staging uses GLOBAL_LOAD_ASYNC_TO_LDS_B128 (double-buffered) when available.
// ---------------------------------------------------------------------------
template<bool STORE_HALF>
__global__ __launch_bounds__(256)
void gemm_wmma(const _Float16* __restrict__ A, int lda, int M,
               const _Float16* __restrict__ Bt, int Kp,
               const float* __restrict__ bias,              // may be nullptr
               float* __restrict__ Cf, _Float16* __restrict__ Ch,
               int ldc, int col0, int do_relu) {
#if HAS_ASYNC_LDS
    __shared__ unsigned sA[2][BM * LROW];    // 2 x  5 KB
    __shared__ unsigned sB[2][256 * LROW];   // 2 x 20 KB
#else
    __shared__ unsigned sA1[BM * LROW];
    __shared__ unsigned sB1[256 * LROW];
#endif

    const int tid   = threadIdx.x;
    const int lane  = tid & 31;
    const int wave  = tid >> 5;
    const int warpM = wave & 3;    // 4 x 16 rows
    const int warpN = wave >> 2;   // 2 x 128 cols
    const int row0  = blockIdx.x * BM;

    // staging coordinates (one 16B chunk of A, four 16B chunks of B per thread)
    const int ar   = tid >> 2;     // 0..63
    const int aseg = tid & 3;
    int arow_g = row0 + ar; if (arow_g > M - 1) arow_g = M - 1;   // clamp
    const _Float16* agp = A + (long long)arow_g * lda + aseg * 8;

    v8f acc[8];
    #pragma unroll
    for (int i = 0; i < 8; ++i) acc[i] = (v8f){};

    const int nK = Kp / BK;
    const int arow  = warpM * 16 + (lane & 15);
    const int ao    = (lane < 16) ? 0 : 4;
    const int nsub  = lane & 15;
    const int koff  = (lane < 16) ? 0 : 4;

#if HAS_ASYNC_LDS
    // issue one K-tile of async global->LDS traffic into buffer `buf`
    auto issue = [&](int buf, int k0) {
        __builtin_amdgcn_global_load_async_to_lds_b128(
            (v4i*)(agp + k0),
            (as3_v4i*)&sA[buf][ar * LROW + aseg * 4], 0, 0);
        #pragma unroll
        for (int j = 0; j < 4; ++j) {
            int idx = tid + j * 256;
            int n = idx >> 2, seg = idx & 3;
            __builtin_amdgcn_global_load_async_to_lds_b128(
                (v4i*)(Bt + (long long)n * Kp + k0 + seg * 8),
                (as3_v4i*)&sB[buf][n * LROW + seg * 4], 0, 0);
        }
    };
    issue(0, 0);
    for (int kt = 0; kt < nK; ++kt) {
        __builtin_amdgcn_s_wait_asynccnt(0);  // this wave's async stores landed
        __syncthreads();                       // everyone's landed; prev reads done
        if (kt + 1 < nK) issue((kt + 1) & 1, (kt + 1) * BK);

        const unsigned* cA = sA[kt & 1];
        const unsigned* cB = sB[kt & 1];
#else
    for (int kt = 0; kt < nK; ++kt) {
        const int k0 = kt * BK;
        __syncthreads();
        {   // direct staged copy (no cross-iteration register pipeline)
            uint4 t0 = *(const uint4*)(agp + k0);
            ((uint4*)(sA1 + ar * LROW))[aseg] = t0;
            #pragma unroll
            for (int j = 0; j < 4; ++j) {
                int idx = tid + j * 256;
                int n = idx >> 2, seg = idx & 3;
                uint4 t = *(const uint4*)(Bt + (long long)n * Kp + k0 + seg * 8);
                ((uint4*)(sB1 + n * LROW))[seg] = t;
            }
        }
        __syncthreads();
        const unsigned* cA = sA1;
        const unsigned* cB = sB1;
#endif
        // A fragment (16x32 f16): lanes0-15 K={0..7,16..23}, lanes16-31 K={8..15,24..31}
        v8u au;
        #pragma unroll
        for (int v = 0; v < 8; ++v) {
            int d = (v < 4 ? v : v + 4) + ao;
            au[v] = cA[arow * LROW + d];
        }
        v16h af = __builtin_bit_cast(v16h, au);

        // 8 B fragments (32x16): lanes0-15 K={0..7,16..23}, lanes16-31 K={8..15,24..31}
        #pragma unroll
        for (int f = 0; f < 8; ++f) {
            v8u bu;
            const int bn = warpN * 128 + f * 16 + nsub;
            #pragma unroll
            for (int v = 0; v < 8; ++v) bu[v] = cB[bn * LROW + koff + (v < 4 ? v : v + 4)];
            v16h bf = __builtin_bit_cast(v16h, bu);
            acc[f] = __builtin_amdgcn_wmma_f32_16x16x32_f16(false, af, false, bf,
                                                            (short)0, acc[f], false, false);
        }
    }

    // C/D layout: VGPR r -> M = r + 8*(lane>=16); N = lane&15
    const int mbase = (lane >> 4) * 8;
    const int ncol  = lane & 15;
    #pragma unroll
    for (int f = 0; f < 8; ++f) {
        const int col = warpN * 128 + f * 16 + ncol;
        const float bv = bias ? bias[col] : 0.0f;
        #pragma unroll
        for (int r = 0; r < 8; ++r) {
            int grow = row0 + warpM * 16 + mbase + r;
            if (grow >= M) continue;
            float v = acc[f][r] + bv;
            if (do_relu) v = fmaxf(v, 0.0f);
            long long o = (long long)grow * ldc + col0 + col;
            if (STORE_HALF) Ch[o] = (_Float16)v;
            else            Cf[o] = v;
        }
    }
}

// ---------------------------------------------------------------------------
// GCN aggregation helpers
// ---------------------------------------------------------------------------
__global__ void deg_init(float* deg, int n) {
    int i = blockIdx.x * blockDim.x + threadIdx.x;
    if (i < n) deg[i] = 1.0f;                       // self-loop
}
__global__ void deg_accum(const int* __restrict__ dst, float* deg, int e) {
    int i = blockIdx.x * blockDim.x + threadIdx.x;
    if (i < e) atomicAdd(&deg[dst[i]], 1.0f);
}
__global__ void deg_to_dinv(float* deg, int n) {
    int i = blockIdx.x * blockDim.x + threadIdx.x;
    if (i < n) deg[i] = rsqrtf(deg[i]);
}
__global__ void selfloop_init(const float* __restrict__ g, const float* __restrict__ dinv,
                              float* __restrict__ agg, long long total) {
    long long i = (long long)blockIdx.x * blockDim.x + threadIdx.x;
    if (i >= total) return;
    float dv = dinv[i >> 8];
    agg[i] = g[i] * dv * dv;
}
__global__ void edge_scatter(const int* __restrict__ src, const int* __restrict__ dst,
                             const float* __restrict__ dinv, const float* __restrict__ g,
                             float* __restrict__ agg, int e) {
    long long t = (long long)blockIdx.x * blockDim.x + threadIdx.x;
    long long ei = t >> 6;
    if (ei >= e) return;
    int q = (int)(t & 63) * 4;
    int s = src[ei], d = dst[ei];
    float coef = dinv[s] * dinv[d];
    const float4 v = *(const float4*)(g + (long long)s * GH + q);
    float* ap = agg + (long long)d * GH + q;
    atomicAdd(ap + 0, v.x * coef);
    atomicAdd(ap + 1, v.y * coef);
    atomicAdd(ap + 2, v.z * coef);
    atomicAdd(ap + 3, v.w * coef);
}
__global__ void bias_relu_to_half(const float* __restrict__ agg, const float* __restrict__ b,
                                  _Float16* __restrict__ out, long long total) {
    long long i = (long long)blockIdx.x * blockDim.x + threadIdx.x;
    if (i >= total) return;
    out[i] = (_Float16)fmaxf(agg[i] + b[i & (GH - 1)], 0.0f);
}
__global__ void bias_relu_inplace(float* __restrict__ agg, const float* __restrict__ b,
                                  long long total) {
    long long i = (long long)blockIdx.x * blockDim.x + threadIdx.x;
    if (i >= total) return;
    agg[i] = fmaxf(agg[i] + b[i & (GH - 1)], 0.0f);
}

// ---------------------------------------------------------------------------
// pooling + head
// ---------------------------------------------------------------------------
__global__ void zerof(float* p, long long n) {
    long long i = (long long)blockIdx.x * blockDim.x + threadIdx.x;
    if (i < n) p[i] = 0.0f;
}
__global__ void cnt_accum(const int* __restrict__ batch, float* cnt, int n) {
    int i = blockIdx.x * blockDim.x + threadIdx.x;
    if (i < n) atomicAdd(&cnt[batch[i]], 1.0f);
}
__global__ void pool_accum(const float* __restrict__ x, const int* __restrict__ batch,
                           float* __restrict__ pooled, int n) {
    long long t = (long long)blockIdx.x * blockDim.x + threadIdx.x;
    long long node = t >> 6;
    if (node >= n) return;
    int q = (int)(t & 63) * 4;
    int b = batch[node];
    const float4 v = *(const float4*)(x + node * GH + q);
    float* p = pooled + (long long)b * GH + q;
    atomicAdd(p + 0, v.x);
    atomicAdd(p + 1, v.y);
    atomicAdd(p + 2, v.z);
    atomicAdd(p + 3, v.w);
}
__global__ void pool_norm(float* pooled, const float* cnt) {
    int i = blockIdx.x * blockDim.x + threadIdx.x;
    if (i < GG * GH) pooled[i] /= fmaxf(cnt[i >> 8], 1.0f);
}
__global__ void head1(const float* __restrict__ pooled, const float* __restrict__ W,
                      const float* __restrict__ b, float* __restrict__ h) {
    int i = blockIdx.x * blockDim.x + threadIdx.x;
    if (i >= GG * GH) return;
    int r = i >> 8, c = i & (GH - 1);
    const float* pr = pooled + r * GH;
    float acc = b[c];
    for (int k = 0; k < GH; ++k) acc += pr[k] * W[k * GH + c];
    h[i] = fmaxf(acc, 0.0f);
}
__global__ void head2(const float* __restrict__ h, const float* __restrict__ W2,
                      const float* __restrict__ b2, float* __restrict__ lg) {
    int i = blockIdx.x * blockDim.x + threadIdx.x;
    if (i >= GG * 2) return;
    int r = i >> 1, o = i & 1;
    const float* hr = h + r * GH;
    float acc = b2[o];
    for (int k = 0; k < GH; ++k) acc += hr[k] * W2[k * 2 + o];
    lg[i] = acc;
}
__global__ void log_softmax2(const float* __restrict__ lg, float* __restrict__ out) {
    int r = blockIdx.x * blockDim.x + threadIdx.x;
    if (r >= GG) return;
    float l0 = lg[2 * r], l1 = lg[2 * r + 1];
    float m = fmaxf(l0, l1);
    float lse = m + logf(expf(l0 - m) + expf(l1 - m));
    out[2 * r] = l0 - lse;
    out[2 * r + 1] = l1 - lse;
}

// ---------------------------------------------------------------------------
// launch
// ---------------------------------------------------------------------------
extern "C" void kernel_launch(void* const* d_in, const int* in_sizes, int n_in,
                              void* d_out, int out_size, void* d_ws, size_t ws_size,
                              hipStream_t stream) {
    const float* content = (const float*)d_in[0];
    const float* bert    = (const float*)d_in[1];
    const float* profile = (const float*)d_in[2];
    const float* spacy   = (const float*)d_in[3];
    const int*   edge    = (const int*)d_in[4];
    const int*   batch   = (const int*)d_in[5];
    const float *Wc = (const float*)d_in[6],  *bc  = (const float*)d_in[7];
    const float *Wb = (const float*)d_in[8],  *bb  = (const float*)d_in[9];
    const float *Wp = (const float*)d_in[10], *bp  = (const float*)d_in[11];
    const float *Ws = (const float*)d_in[12], *bs  = (const float*)d_in[13];
    const float *Wg1 = (const float*)d_in[14], *bg1 = (const float*)d_in[15];
    const float *Wg2 = (const float*)d_in[16], *bg2 = (const float*)d_in[17];
    const float *Wl1 = (const float*)d_in[18], *bl1 = (const float*)d_in[19];
    const float *Wl2 = (const float*)d_in[20], *bl2 = (const float*)d_in[21];
    const int* src = edge;
    const int* dst = edge + GE;
    float* out = (float*)d_out;

    // ---- workspace layout ----
    char* w = (char*)d_ws;
    auto alloc = [&](size_t bytes) -> void* {
        void* p = (void*)w;
        w += (bytes + 255) & ~(size_t)255;
        return p;
    };
    auto algn = [](size_t b) { return (b + 255) & ~(size_t)255; };

    _Float16* h1h = (_Float16*)alloc((size_t)GN * 1024 * 2);     // concat, f16

    // union region: padded f16 features (branch phase)  VS  gcn-phase buffers
    size_t featBytes = algn((size_t)GN*320*2) + algn((size_t)GN*768*2)
                     + algn((size_t)GN*32*2)  + algn((size_t)GN*320*2);
    size_t gcnBytes  = algn((size_t)GN*GH*4) + algn((size_t)GN*GH*4)
                     + algn((size_t)GN*GH*2);
    char* blob = (char*)alloc(featBytes > gcnBytes ? featBytes : gcnBytes);
    _Float16* cF16 = (_Float16*)blob;
    _Float16* bF16 = (_Float16*)(blob + algn((size_t)GN*320*2));
    _Float16* pF16 = (_Float16*)(blob + algn((size_t)GN*320*2) + algn((size_t)GN*768*2));
    _Float16* sF16 = (_Float16*)(blob + algn((size_t)GN*320*2) + algn((size_t)GN*768*2)
                                      + algn((size_t)GN*32*2));
    float*    gbuf = (float*)blob;
    float*    agg  = (float*)(blob + algn((size_t)GN*GH*4));
    _Float16* x2h  = (_Float16*)(blob + 2 * algn((size_t)GN*GH*4));

    float*    dinv  = (float*)   alloc((size_t)GN * 4);
    float*    pooled= (float*)   alloc((size_t)GG * GH * 4);
    float*    cnt   = (float*)   alloc((size_t)GG * 4);
    float*    hbuf  = (float*)   alloc((size_t)GG * GH * 4);
    float*    lgbuf = (float*)   alloc((size_t)GG * 2 * 4);
    _Float16* WcT  = (_Float16*)alloc((size_t)256 * 320 * 2);
    _Float16* WbT  = (_Float16*)alloc((size_t)256 * 768 * 2);
    _Float16* WpT  = (_Float16*)alloc((size_t)256 * 32 * 2);
    _Float16* WsT  = (_Float16*)alloc((size_t)256 * 320 * 2);
    _Float16* Wg1T = (_Float16*)alloc((size_t)256 * 1024 * 2);
    _Float16* Wg2T = (_Float16*)alloc((size_t)256 * 256 * 2);

    const int TB = 256;
    auto cdiv = [](long long a, long long b) { return (int)((a + b - 1) / b); };

    // ---- weights -> transposed padded f16 ----
    cvt_transpose<<<cdiv(256LL*320, TB), TB, 0, stream>>>(Wc, WcT, 310, 256, 320);
    cvt_transpose<<<cdiv(256LL*768, TB), TB, 0, stream>>>(Wb, WbT, 768, 256, 768);
    cvt_transpose<<<cdiv(256LL*32,  TB), TB, 0, stream>>>(Wp, WpT, 10,  256, 32);
    cvt_transpose<<<cdiv(256LL*320, TB), TB, 0, stream>>>(Ws, WsT, 300, 256, 320);
    cvt_transpose<<<cdiv(256LL*1024,TB), TB, 0, stream>>>(Wg1, Wg1T, 1024, 256, 1024);
    cvt_transpose<<<cdiv(256LL*256, TB), TB, 0, stream>>>(Wg2, Wg2T, 256, 256, 256);

    // ---- features -> padded f16 ----
    cast_pad_f16<<<cdiv((long long)GN*320, TB), TB, 0, stream>>>(content, cF16, 310, 320, GN);
    cast_pad_f16<<<cdiv((long long)GN*768, TB), TB, 0, stream>>>(bert,    bF16, 768, 768, GN);
    cast_pad_f16<<<cdiv((long long)GN*32,  TB), TB, 0, stream>>>(profile, pF16, 10,  32,  GN);
    cast_pad_f16<<<cdiv((long long)GN*320, TB), TB, 0, stream>>>(spacy,   sF16, 300, 320, GN);

    // ---- branch linears + relu -> h1h [N,1024] f16 ----
    const int gx = cdiv(GN, BM);
    gemm_wmma<true><<<gx, TB, 0, stream>>>(cF16, 320, GN, WcT, 320, bc,
                                           nullptr, h1h, 1024, 0,   1);
    gemm_wmma<true><<<gx, TB, 0, stream>>>(bF16, 768, GN, WbT, 768, bb,
                                           nullptr, h1h, 1024, 256, 1);
    gemm_wmma<true><<<gx, TB, 0, stream>>>(pF16, 32,  GN, WpT, 32,  bp,
                                           nullptr, h1h, 1024, 512, 1);
    gemm_wmma<true><<<gx, TB, 0, stream>>>(sF16, 320, GN, WsT, 320, bs,
                                           nullptr, h1h, 1024, 768, 1);

    // ---- degrees / dinv (shared by both convs) ----
    deg_init<<<cdiv(GN, TB), TB, 0, stream>>>(dinv, GN);
    deg_accum<<<cdiv(GE, TB), TB, 0, stream>>>(dst, dinv, GE);
    deg_to_dinv<<<cdiv(GN, TB), TB, 0, stream>>>(dinv, GN);

    const long long totNH = (long long)GN * GH;

    // ---- GCN conv 1 ----  (h1h read; gbuf overlays now-dead feature copies)
    gemm_wmma<false><<<gx, TB, 0, stream>>>(h1h, 1024, GN, Wg1T, 1024, nullptr,
                                            gbuf, nullptr, GH, 0, 0);
    selfloop_init<<<cdiv(totNH, TB), TB, 0, stream>>>(gbuf, dinv, agg, totNH);
    edge_scatter<<<cdiv((long long)GE * 64, TB), TB, 0, stream>>>(src, dst, dinv, gbuf, agg, GE);
    bias_relu_to_half<<<cdiv(totNH, TB), TB, 0, stream>>>(agg, bg1, x2h, totNH);

    // ---- GCN conv 2 ----
    gemm_wmma<false><<<gx, TB, 0, stream>>>(x2h, GH, GN, Wg2T, GH, nullptr,
                                            gbuf, nullptr, GH, 0, 0);
    selfloop_init<<<cdiv(totNH, TB), TB, 0, stream>>>(gbuf, dinv, agg, totNH);
    edge_scatter<<<cdiv((long long)GE * 64, TB), TB, 0, stream>>>(src, dst, dinv, gbuf, agg, GE);
    bias_relu_inplace<<<cdiv(totNH, TB), TB, 0, stream>>>(agg, bg2, totNH);

    // ---- global mean pool ----
    zerof<<<cdiv(GG * GH, TB), TB, 0, stream>>>(pooled, GG * GH);
    zerof<<<1, GG, 0, stream>>>(cnt, GG);
    cnt_accum<<<cdiv(GN, TB), TB, 0, stream>>>(batch, cnt, GN);
    pool_accum<<<cdiv((long long)GN * 64, TB), TB, 0, stream>>>(agg, batch, pooled, GN);
    pool_norm<<<cdiv(GG * GH, TB), TB, 0, stream>>>(pooled, cnt);

    // ---- head ----
    head1<<<cdiv(GG * GH, TB), TB, 0, stream>>>(pooled, Wl1, bl1, hbuf);
    head2<<<1, 128, 0, stream>>>(hbuf, Wl2, bl2, lgbuf);
    log_softmax2<<<1, GG, 0, stream>>>(lgbuf, out);
}
